// Zeroshot_41996190220784
// MI455X (gfx1250) — matrix operand order repaired and checked by reading, not compiled, not run on hardware
//
#include <hip/hip_runtime.h>

typedef __attribute__((ext_vector_type(16))) __bf16 v16bf;
typedef __attribute__((ext_vector_type(8)))  __bf16 v8bf;
typedef __attribute__((ext_vector_type(8)))  float  v8f;
typedef __attribute__((ext_vector_type(4)))  unsigned v4u;
typedef __attribute__((ext_vector_type(8)))  int      v8i;
typedef __attribute__((ext_vector_type(4)))  int      v4i;
typedef unsigned long long u64;

#if __has_builtin(__builtin_amdgcn_tensor_load_to_lds) && \
    __has_builtin(__builtin_amdgcn_s_wait_tensorcnt)
#define USE_TDM 1
#else
#define USE_TDM 0
#endif

// Problem dims
constexpr int Bk = 32, Sk = 512, Nk = 512, Ck = 1024, Ak = 312, Hk = 900;
// Padded dims: K mult-of-32 (WMMA), N mult-of-128 (block tile)
constexpr int KA = 320;    // A padded (K of GEMM1)
constexpr int HN = 1024;   // H padded (N of GEMM1, K of GEMM2)
constexpr int ROWS = Bk * Nk;     // 16384 (= Bk*Sk as well)
constexpr int LDSP = 40;          // LDS row pitch in bf16 (80 B, conflict-free)
constexpr int TILE_BYTES = 128 * LDSP * 2;  // one LDS tile buffer

__device__ __forceinline__ __bf16 f2bf(float f) {
  unsigned u = __float_as_uint(f);
  unsigned r = (u + 0x7fffu + ((u >> 16) & 1u)) >> 16;  // RNE
  unsigned short h = (unsigned short)r;
  return __builtin_bit_cast(__bf16, h);
}
__device__ __forceinline__ float bf2f(__bf16 h) {
  unsigned short u = __builtin_bit_cast(unsigned short, h);
  return __uint_as_float(((unsigned)u) << 16);
}
__device__ __forceinline__ unsigned fkey(float f) {  // order-preserving f32->u32
  unsigned u = __float_as_uint(f);
  return (u & 0x80000000u) ? ~u : (u | 0x80000000u);
}
__device__ __forceinline__ u64 shfl_xor_u64(u64 v, int m) {
  int lo = __shfl_xor((int)(unsigned)v, m, 32);
  int hi = __shfl_xor((int)(unsigned)(v >> 32), m, 32);
  return ((u64)(unsigned)hi << 32) | (unsigned)lo;
}

#if USE_TDM
// Issue one TDM 2-D tile load: 128 rows x 32 bf16 from row-major [.., ld] into
// LDS at ldsByteAddr with 80B row pitch produced by the TDM pad feature
// (pad_interval code 3 = every 16 dwords, pad_amount code 3 = 4 dwords).
__device__ __forceinline__ void tdm_load_tile(const __bf16* g, int ldElems,
                                              unsigned ldsByteAddr) {
  u64 ga = (u64)(size_t)g;
  v4u g0;
  g0[0] = 1u;                                   // count=1 (valid), user mode
  g0[1] = ldsByteAddr;                          // lds_addr
  g0[2] = (unsigned)(ga & 0xffffffffu);         // global_addr[31:0]
  g0[3] = (unsigned)((ga >> 32) & 0x01ffffffu)  // global_addr[56:32]
          | (2u << 30);                         // type=2 ("image")
  unsigned td0 = 0x7fffffffu, td1 = 0x7fffffffu;  // huge tensor dims (in-bounds)
  u64 st0 = (u64)(unsigned)ldElems;               // tensor_dim0_stride (elems)
  v8i g1;
  g1[0] = (int)((1u << 16)      // data_size=1 -> 2 bytes
              | (1u << 20)      // pad_enable
              | (3u << 22)      // pad_interval: 16 dwords (64 B tile row)
              | (3u << 25));    // pad_amount: 4 dwords (16 B) -> 80 B pitch
  g1[1] = (int)((td0 & 0xffffu) << 16);                         // td0[15:0]
  g1[2] = (int)(((td0 >> 16) & 0xffffu) | ((td1 & 0xffffu) << 16));
  g1[3] = (int)(((td1 >> 16) & 0xffffu) | (32u << 16));         // tile_dim0=32
  g1[4] = (int)128;                                             // tile_dim1=128
  g1[5] = (int)(st0 & 0xffffffffu);
  g1[6] = (int)((st0 >> 32) & 0xffffu);                         // stride0 hi16
  g1[7] = 0;
  v4i gz = {0, 0, 0, 0};
#if defined(__clang_major__) && __clang_major__ >= 23
  v8i gz8 = {0, 0, 0, 0, 0, 0, 0, 0};
  __builtin_amdgcn_tensor_load_to_lds(g0, g1, gz, gz, gz8, 0);
#else
  __builtin_amdgcn_tensor_load_to_lds(g0, g1, gz, gz, 0);
#endif
}
#endif

// ---------- conversion kernels ----------
__global__ void zs_cvt_pad(const float* __restrict__ src, __bf16* __restrict__ dst,
                           int R, int Ks, int Kd) {
  long long idx = (long long)blockIdx.x * blockDim.x + threadIdx.x;
  long long tot = (long long)R * Kd;
  if (idx >= tot) return;
  int k = (int)(idx % Kd);
  long long r = idx / Kd;
  __bf16 v = __builtin_bit_cast(__bf16, (unsigned short)0);
  if (k < Ks) v = f2bf(src[r * Ks + k]);
  dst[idx] = v;
}
__global__ void zs_cvt_tpad(const float* __restrict__ src, __bf16* __restrict__ dst,
                            int Ks, int Ns, int Kd, int Nd) {
  long long idx = (long long)blockIdx.x * blockDim.x + threadIdx.x;
  long long tot = (long long)Nd * Kd;
  if (idx >= tot) return;
  int k = (int)(idx % Kd);
  int n = (int)(idx / Kd);
  __bf16 v = __builtin_bit_cast(__bf16, (unsigned short)0);
  if (k < Ks && n < Ns) v = f2bf(src[(long long)k * Ns + n]);
  dst[idx] = v;
}

// one K=32 step: 2 A sub-tiles x 4 B sub-tiles -> 8 WMMAs
__device__ __forceinline__ void mma_step(const __bf16* lA, const __bf16* lB,
                                         int wm, int wn, int r, int half,
                                         v8f acc[2][4]) {
  v16bf av[2], bv[4];
#pragma unroll
  for (int i = 0; i < 2; ++i) {
    // A 16x32 operand: lanes 0-15 K{0..7,16..23}; lanes 16-31 K{8..15,24..31}
    int row = wm + i * 16 + r;
    v8bf lo = *(const v8bf*)&lA[row * LDSP + half * 8];
    v8bf hi = *(const v8bf*)&lA[row * LDSP + 16 + half * 8];
    av[i] = __builtin_shufflevector(lo, hi, 0,1,2,3,4,5,6,7,8,9,10,11,12,13,14,15);
  }
#pragma unroll
  for (int j = 0; j < 4; ++j) {
    // B 32x16 operand: lanes 0-15 K{0..7,16..23}; lanes 16-31 K{8..15,24..31}
    int nr = wn + j * 16 + r;
    v8bf lo = *(const v8bf*)&lB[nr * LDSP + half * 8];
    v8bf hi = *(const v8bf*)&lB[nr * LDSP + 16 + half * 8];
    bv[j] = __builtin_shufflevector(lo, hi, 0,1,2,3,4,5,6,7,8,9,10,11,12,13,14,15);
  }
#pragma unroll
  for (int i = 0; i < 2; ++i)
#pragma unroll
    for (int j = 0; j < 4; ++j)
      acc[i][j] = __builtin_amdgcn_wmma_f32_16x16x32_bf16(
          false, av[i], false, bv[j], (short)0, acc[i][j], false, false);
}

// ---------- main WMMA GEMM: C = A[M,K] * Bt[N,K]^T ----------
// 256 threads = 8 waves; block tile 128x128; wave tile 32x64.
// TDM double-buffer: wave 0 DMAs tile k+1 while all waves run WMMA on tile k.
// mode 0/1: out_bf16 = relu(acc + bias[n<biasN])
// mode 2  : score = e2[b*Nk+n] - 2*acc; packed-u64 atomicMin argmin per row
__global__ __launch_bounds__(256) void zs_gemm_bf16(
    const __bf16* __restrict__ A, int lda, long long strideA,
    const __bf16* __restrict__ Bt, int ldb, long long strideB,
    int K, int mTiles, int nTiles, int mode,
    const float* __restrict__ bias, int biasN,
    __bf16* __restrict__ outBf, int ldo,
    const float* __restrict__ e2, u64* __restrict__ minpack) {
  __shared__ __bf16 ldsA[2][128 * LDSP];
  __shared__ __bf16 ldsB[2][128 * LDSP];

  int tilesPerBatch = mTiles * nTiles;
  int b = blockIdx.x / tilesPerBatch;
  int t = blockIdx.x - b * tilesPerBatch;
  int mBase = (t / nTiles) * 128;
  int nBase = (t % nTiles) * 128;

  const __bf16* Ab = A + (long long)b * strideA + (long long)mBase * lda;
  const __bf16* Bb = Bt + (long long)b * strideB + (long long)nBase * ldb;

  int tid  = threadIdx.x;
  int lane = tid & 31;
  int w    = tid >> 5;
  int half = lane >> 4;
  int r    = lane & 15;
  int wm   = (w & 3) * 32;   // 4 wave rows
  int wn   = (w >> 2) * 64;  // 2 wave cols

  v8f acc[2][4];
#pragma unroll
  for (int i = 0; i < 2; ++i)
#pragma unroll
    for (int j = 0; j < 4; ++j)
#pragma unroll
      for (int v = 0; v < 8; ++v) acc[i][j][v] = 0.f;

  int ksteps = K / 32;

#if USE_TDM
  // LDS flat pointers carry the LDS byte address in the low 32 bits (ISA 10.2)
  unsigned ldsAbase = (unsigned)(size_t)(void*)&ldsA[0][0];
  unsigned ldsBbase = (unsigned)(size_t)(void*)&ldsB[0][0];
  if (w == 0) {
    tdm_load_tile(Ab, lda, ldsAbase);
    tdm_load_tile(Bb, ldb, ldsBbase);
  }
  int cur = 0;
  for (int ks = 0; ks < ksteps; ++ks) {
    if (w == 0) __builtin_amdgcn_s_wait_tensorcnt(0);
    __syncthreads();  // publish buf[cur]
    if (w == 0 && ks + 1 < ksteps) {  // DMA next tile under the WMMAs
      unsigned off = (unsigned)((cur ^ 1) * TILE_BYTES);
      tdm_load_tile(Ab + (ks + 1) * 32, lda, ldsAbase + off);
      tdm_load_tile(Bb + (ks + 1) * 32, ldb, ldsBbase + off);
    }
    mma_step(&ldsA[cur][0], &ldsB[cur][0], wm, wn, r, half, acc);
    __syncthreads();  // all waves done reading buf[cur]
    cur ^= 1;
  }
#else
  // fallback: manual global->LDS staging, single buffer
  int arow = tid >> 1;
  int acol = (tid & 1) * 16;
  for (int ks = 0; ks < ksteps; ++ks) {
    int k0 = ks * 32;
    const __bf16* gA = Ab + (long long)arow * lda + k0 + acol;
    *(v8bf*)&ldsA[0][arow * LDSP + acol]     = *(const v8bf*)(gA);
    *(v8bf*)&ldsA[0][arow * LDSP + acol + 8] = *(const v8bf*)(gA + 8);
    const __bf16* gB = Bb + (long long)arow * ldb + k0 + acol;
    *(v8bf*)&ldsB[0][arow * LDSP + acol]     = *(const v8bf*)(gB);
    *(v8bf*)&ldsB[0][arow * LDSP + acol + 8] = *(const v8bf*)(gB + 8);
    __syncthreads();
    mma_step(&ldsA[0][0], &ldsB[0][0], wm, wn, r, half, acc);
    __syncthreads();
  }
#endif

  // C/D layout: VGPR v -> M = base + v + 8*half, N = base + (lane&15)
  if (mode != 2) {
#pragma unroll
    for (int i = 0; i < 2; ++i)
#pragma unroll
      for (int j = 0; j < 4; ++j) {
        int n = nBase + wn + j * 16 + r;
        float bval = (n < biasN) ? bias[n] : 0.f;
#pragma unroll
        for (int v = 0; v < 8; ++v) {
          int m = mBase + wm + i * 16 + half * 8 + v;
          float val = fmaxf(acc[i][j][v] + bval, 0.f);
          outBf[(long long)m * ldo + n] = f2bf(val);
        }
      }
  } else {
#pragma unroll
    for (int i = 0; i < 2; ++i) {
#pragma unroll
      for (int v = 0; v < 8; ++v) {
        int m = mBase + wm + i * 16 + half * 8 + v;  // row (s) within batch
        u64 best = ~0ull;
#pragma unroll
        for (int j = 0; j < 4; ++j) {
          int n = nBase + wn + j * 16 + r;
          float sc = e2[(long long)b * Nk + n] - 2.0f * acc[i][j][v];
          u64 key = ((u64)fkey(sc) << 32) | (unsigned)n;
          if (key < best) best = key;
        }
#pragma unroll
        for (int d = 1; d < 16; d <<= 1) {  // min over 16 lanes of this half
          u64 o = shfl_xor_u64(best, d);
          if (o < best) best = o;
        }
        if (r == 0) atomicMin(&minpack[(long long)b * Sk + m], best);
      }
    }
  }
}

// ---------- e2 = rowwise sum of squares (deterministic, no float atomics) ----
__global__ void zs_rowsumsq(const __bf16* __restrict__ emb, float* __restrict__ e2,
                            int rows, int cols) {
  int row = blockIdx.x * (blockDim.x >> 5) + (threadIdx.x >> 5);
  int lane = threadIdx.x & 31;
  if (row >= rows) return;
  float s = 0.f;
  for (int c = lane; c < cols; c += 32) {
    float v = bf2f(emb[(long long)row * cols + c]);
    s += v * v;
  }
#pragma unroll
  for (int d = 16; d; d >>= 1) s += __shfl_xor(s, d, 32);
  if (lane == 0) e2[row] = s;
}

__global__ void zs_init(u64* __restrict__ minpack, int* __restrict__ counter, int n) {
  int i = blockIdx.x * blockDim.x + threadIdx.x;
  if (i < n) minpack[i] = ~0ull;
  if (i == 0) *counter = 0;
}

__global__ void zs_finalize(const u64* __restrict__ minpack,
                            const long long* __restrict__ att_label,
                            const long long* __restrict__ x_label,
                            float* __restrict__ out, int* __restrict__ counter) {
  int i = blockIdx.x * blockDim.x + threadIdx.x;
  if (i >= Bk * Sk) return;
  int b = i >> 9;
  unsigned n = (unsigned)(minpack[i] & 0xffffffffull);
  long long p = att_label[(long long)b * Nk + n];
  out[i] = (float)p;
  if (p == x_label[i]) atomicAdd(counter, 1);
}

__global__ void zs_write_correct(const int* __restrict__ counter, float* __restrict__ out) {
  out[Bk * Sk] = (float)(*counter);
}

extern "C" void kernel_launch(void* const* d_in, const int* in_sizes, int n_in,
                              void* d_out, int out_size, void* d_ws, size_t ws_size,
                              hipStream_t stream) {
  const float*     x   = (const float*)d_in[0];
  const long long* xl  = (const long long*)d_in[1];
  const float*     att = (const float*)d_in[2];
  const long long* al  = (const long long*)d_in[3];
  const float*     W1  = (const float*)d_in[4];
  const float*     b1  = (const float*)d_in[5];
  const float*     W2  = (const float*)d_in[6];
  const float*     b2  = (const float*)d_in[7];
  float* out = (float*)d_out;

  char* p = (char*)d_ws;
  auto carve = [&](size_t bytes) -> char* {
    char* q = p;
    p += (bytes + 255) & ~(size_t)255;
    return q;
  };
  __bf16* attb = (__bf16*)carve((size_t)ROWS * KA * 2);
  __bf16* xb   = (__bf16*)carve((size_t)ROWS * Ck * 2);
  __bf16* W1t  = (__bf16*)carve((size_t)HN * KA * 2);
  __bf16* W2t  = (__bf16*)carve((size_t)Ck * HN * 2);
  __bf16* hb   = (__bf16*)carve((size_t)ROWS * HN * 2);
  __bf16* embb = (__bf16*)carve((size_t)ROWS * Ck * 2);
  float*  e2   = (float*)carve((size_t)ROWS * 4);
  u64*    mp   = (u64*)carve((size_t)(Bk * Sk) * 8);
  int*    cnt  = (int*)carve(256);

  dim3 blk(256);
  zs_cvt_pad<<<((size_t)ROWS * KA + 255) / 256, blk, 0, stream>>>(att, attb, ROWS, Ak, KA);
  zs_cvt_pad<<<((size_t)ROWS * Ck + 255) / 256, blk, 0, stream>>>(x, xb, ROWS, Ck, Ck);
  zs_cvt_tpad<<<((size_t)HN * KA + 255) / 256, blk, 0, stream>>>(W1, W1t, Ak, Hk, KA, HN);
  zs_cvt_tpad<<<((size_t)Ck * HN + 255) / 256, blk, 0, stream>>>(W2, W2t, Hk, Ck, HN, Ck);

  // GEMM1: h = relu(att @ W1 + b1)  [16384 x 1024], K=320
  zs_gemm_bf16<<<(ROWS / 128) * (HN / 128), blk, 0, stream>>>(
      attb, KA, 0, W1t, KA, 0, KA, ROWS / 128, HN / 128, 0,
      b1, Hk, hb, HN, nullptr, nullptr);
  // GEMM2: emb = relu(h @ W2 + b2)  [16384 x 1024], K=1024
  zs_gemm_bf16<<<(ROWS / 128) * (Ck / 128), blk, 0, stream>>>(
      hb, HN, 0, W2t, HN, 0, HN, ROWS / 128, Ck / 128, 1,
      b2, Ck, embb, Ck, nullptr, nullptr);

  zs_rowsumsq<<<ROWS / 8, blk, 0, stream>>>(embb, e2, ROWS, Ck);
  zs_init<<<(Bk * Sk + 255) / 256, blk, 0, stream>>>(mp, cnt, Bk * Sk);

  // GEMM3 (batched over B): xe = x_b @ emb_b^T, fused e2[n]-2*xe + u64 argmin
  zs_gemm_bf16<<<Bk * (Sk / 128) * (Nk / 128), blk, 0, stream>>>(
      xb, Ck, (long long)Sk * Ck, embb, Ck, (long long)Nk * Ck,
      Ck, Sk / 128, Nk / 128, 2,
      nullptr, 0, nullptr, 0, e2, mp);

  zs_finalize<<<(Bk * Sk + 255) / 256, blk, 0, stream>>>(mp, al, xl, out, cnt);
  zs_write_correct<<<1, 1, 0, stream>>>(cnt, out);
}
